// WGCC_weight_42640435315042
// MI455X (gfx1250) — matrix operand order repaired and checked by reading, not compiled, not run on hardware
//
#include <hip/hip_runtime.h>
#include <hip/hip_fp16.h>

typedef _Float16 f16;
typedef __attribute__((ext_vector_type(16))) _Float16 v16h;
typedef __attribute__((ext_vector_type(8)))  _Float16 v8h;
typedef __attribute__((ext_vector_type(8)))  float    v8f;

#define WMMA_F16(A, B, C) \
  __builtin_amdgcn_wmma_f32_16x16x32_f16(false, (A), false, (B), (short)0, (C), false, false)

#define NTOK   2048
#define HDIM   128
#define NBATCH 8
#define NROWS  (NBATCH * NTOK)          // 16384
#define INV_T  0.08838834764831845f     // 1/sqrt(128)

// ---------------------------------------------------------------------------
// Kernel 1: res = q@W1 ; qh=(res@Wq)/T ; kh=res@Wk ; vh_t=(res@Wv)^T  (f16 ws)
// Block = 128 threads (4 waves), each wave owns 16 rows. Grid = 256.
// vh is written TRANSPOSED per batch: vh_t[(b*128 + d)*2048 + token], which
// the C/D layout gives us for free as packed v8h stores (8 tokens per lane).
// ---------------------------------------------------------------------------
__global__ __launch_bounds__(128) void proj_kernel(
    const float* __restrict__ q,
    const float* __restrict__ W1, const float* __restrict__ Wq,
    const float* __restrict__ Wk, const float* __restrict__ Wv,
    float* __restrict__ out,       // [16384,256] concat output
    float* __restrict__ res_out,   // [16384,128]
    f16* __restrict__ qh_h, f16* __restrict__ kh_h, f16* __restrict__ vh_t)
{
  __shared__ __attribute__((aligned(32))) f16 wt[HDIM * HDIM];     // W^T f16
  __shared__ __attribute__((aligned(32))) f16 stage[4][16 * HDIM]; // res per wave

  const int tid  = threadIdx.x;
  const int wave = tid >> 5;
  const int lane = tid & 31;
  const int h    = lane >> 4;   // half-wave
  const int ln   = lane & 15;
  const int row0 = blockIdx.x * 64 + wave * 16;   // global row (flat b*2048+tok)
  const int b    = row0 >> 11;
  const int tok0 = row0 & (NTOK - 1);

  // ---- load q tile as A-fragments (4 k-steps of 32), convert to f16
  v16h qfrag[4];
  {
    const float* qr = q + (size_t)(row0 + ln) * HDIM;
    #pragma unroll
    for (int s = 0; s < 4; ++s) {
      #pragma unroll
      for (int e = 0; e < 8; ++e) {
        qfrag[s][e]     = (f16)qr[32 * s + 16 * h + e];
        qfrag[s][e + 8] = (f16)qr[32 * s + 16 * h + 8 + e];
      }
    }
  }

  auto stageW = [&](const float* W) {
    for (int i = tid; i < HDIM * HDIM; i += 128) {
      int k = i >> 7, n = i & 127;
      wt[n * HDIM + k] = (f16)W[i];   // transposed so B-frags are contiguous
    }
  };

  auto gemm = [&](const v16h* af, v8f* acc) {
    v8f z = {};
    #pragma unroll
    for (int n = 0; n < 8; ++n) acc[n] = z;
    #pragma unroll
    for (int s = 0; s < 4; ++s) {
      #pragma unroll
      for (int n = 0; n < 8; ++n) {
        const v16h bf = *(const v16h*)&wt[(16 * n + ln) * HDIM + 32 * s + 16 * h];
        acc[n] = WMMA_F16(af[s], bf, acc[n]);
      }
    }
  };

  v8f acc[8];

  // ---- GEMM 1: res = q @ W1
  stageW(W1);
  __syncthreads();
  gemm(qfrag, acc);

  f16* st = stage[wave];
  #pragma unroll
  for (int n = 0; n < 8; ++n) {
    #pragma unroll
    for (int r = 0; r < 8; ++r) {
      float v = acc[n][r];
      int   m = r + 8 * h;
      int   c = 16 * n + ln;
      size_t grow = (size_t)row0 + m;
      out[grow * 256 + 128 + c] = v;   // concat right half
      res_out[grow * 128 + c]   = v;   // second tuple output
      st[m * HDIM + c]          = (f16)v;
    }
  }

  // ---- reload res as A-fragments (wave-local LDS, DS ops are in-order)
  v16h rfrag[4];
  #pragma unroll
  for (int s = 0; s < 4; ++s) {
    ((v8h*)&rfrag[s])[0] = *(const v8h*)&st[ln * HDIM + 32 * s + 16 * h];
    ((v8h*)&rfrag[s])[1] = *(const v8h*)&st[ln * HDIM + 32 * s + 16 * h + 8];
  }

  // ---- GEMM 2: qh = (res @ Wq) / T
  __syncthreads();
  stageW(Wq);
  __syncthreads();
  gemm(rfrag, acc);
  #pragma unroll
  for (int n = 0; n < 8; ++n)
    #pragma unroll
    for (int r = 0; r < 8; ++r)
      qh_h[((size_t)row0 + r + 8 * h) * HDIM + 16 * n + ln] = (f16)(acc[n][r] * INV_T);

  // ---- GEMM 3: kh = res @ Wk   (row-major for contiguous K^T fragments)
  __syncthreads();
  stageW(Wk);
  __syncthreads();
  gemm(rfrag, acc);
  #pragma unroll
  for (int n = 0; n < 8; ++n)
    #pragma unroll
    for (int r = 0; r < 8; ++r)
      kh_h[((size_t)row0 + r + 8 * h) * HDIM + 16 * n + ln] = (f16)acc[n][r];

  // ---- GEMM 4: vh_t = (res @ Wv)^T, packed 8-token v8h stores per lane
  __syncthreads();
  stageW(Wv);
  __syncthreads();
  gemm(rfrag, acc);
  #pragma unroll
  for (int n = 0; n < 8; ++n) {
    v8h pk;
    #pragma unroll
    for (int r = 0; r < 8; ++r) pk[r] = (f16)acc[n][r];
    int d = 16 * n + ln;
    *(v8h*)&vh_t[((size_t)b * HDIM + d) * NTOK + tok0 + 8 * h] = pk;
  }
}

// ---------------------------------------------------------------------------
// Kernel 2: flash attention with adj mask, double-buffered LDS staging.
// Block = 128 threads (4 waves), wave = 16 query rows. Grid = (32, 8).
// ---------------------------------------------------------------------------
__global__ __launch_bounds__(128) void attn_kernel(
    const f16* __restrict__ qh_h, const f16* __restrict__ kh_h,
    const f16* __restrict__ vh_t, const float* __restrict__ adj,
    float* __restrict__ out)
{
  __shared__ __attribute__((aligned(32))) f16 kh_lds[2][64 * HDIM]; // [key][d]
  __shared__ __attribute__((aligned(32))) f16 vt_lds[2][HDIM * 64]; // [d][key]
  __shared__ __attribute__((aligned(32))) f16 p_lds[4][16 * 64];    // per wave

  const int tid  = threadIdx.x;
  const int wave = tid >> 5;
  const int lane = tid & 31;
  const int h    = lane >> 4;
  const int ln   = lane & 15;
  const int b    = blockIdx.y;
  const int q0   = blockIdx.x * 64 + wave * 16;       // query row in batch
  const size_t qrow = (size_t)b * NTOK + q0;

  const f16*   khg  = kh_h + (size_t)b * NTOK * HDIM;      // [tok][d]
  const f16*   vtg  = vh_t + (size_t)b * HDIM * NTOK;      // [d][tok]
  const float* adjb = adj + (size_t)b * NTOK * NTOK;

  // ---- Q A-fragments, resident for the whole key loop
  v16h qfrag[4];
  {
    const f16* qr = qh_h + (qrow + ln) * HDIM;
    #pragma unroll
    for (int s = 0; s < 4; ++s) {
      ((v8h*)&qfrag[s])[0] = *(const v8h*)&qr[32 * s + 16 * h];
      ((v8h*)&qfrag[s])[1] = *(const v8h*)&qr[32 * s + 16 * h + 8];
    }
  }

  v8f z = {};
  v8f o_acc[8];
  #pragma unroll
  for (int n = 0; n < 8; ++n) o_acc[n] = z;
  float mrow[8], lrow[8];
  #pragma unroll
  for (int r = 0; r < 8; ++r) { mrow[r] = -3.0e38f; lrow[r] = 0.f; }

  // ---- double-buffered staging helpers (8 x v8h each = 8192 f16 per tile)
  v8h kreg[8], vreg[8];
  auto load_tile = [&](int j0) {
    #pragma unroll
    for (int c = 0; c < 8; ++c) {
      int i = tid * 8 + c * 1024;
      kreg[c] = *(const v8h*)&khg[(size_t)j0 * HDIM + i];        // contiguous
      int d = i >> 6, jj = i & 63;
      vreg[c] = *(const v8h*)&vtg[(size_t)d * NTOK + j0 + jj];   // strided rows
    }
  };
  auto store_tile = [&](int buf) {
    #pragma unroll
    for (int c = 0; c < 8; ++c) {
      int i = tid * 8 + c * 1024;
      *(v8h*)&kh_lds[buf][i] = kreg[c];
      *(v8h*)&vt_lds[buf][i] = vreg[c];
    }
  };

  load_tile(0);
  store_tile(0);
  int cur = 0;

  for (int jt = 0; jt < NTOK / 64; ++jt) {
    const int j0 = jt * 64;
    __syncthreads();                       // buffer `cur` ready for all waves
    if (jt + 1 < NTOK / 64) load_tile(j0 + 64);   // overlap with compute

    // ---- S = Q K^T : 4 column tiles of 16 keys, 4 k-steps over H=128
    v8f s_acc[4];
    #pragma unroll
    for (int t = 0; t < 4; ++t) s_acc[t] = z;
    #pragma unroll
    for (int s = 0; s < 4; ++s) {
      #pragma unroll
      for (int t = 0; t < 4; ++t) {
        const v16h kf =
            *(const v16h*)&kh_lds[cur][(16 * t + ln) * HDIM + 32 * s + 16 * h];
        s_acc[t] = WMMA_F16(qfrag[s], kf, s_acc[t]);
      }
    }

    // ---- adj mask (the dominant HBM stream, fused here)
    #pragma unroll
    for (int t = 0; t < 4; ++t) {
      #pragma unroll
      for (int r = 0; r < 8; ++r) {
        float a = adjb[(size_t)(q0 + r + 8 * h) * NTOK + j0 + 16 * t + ln];
        s_acc[t][r] = (a == 0.f) ? -1.0e9f : s_acc[t][r];
      }
    }

    // ---- online softmax, rows live in 16-lane groups (C/D layout)
    float alpha[8];
    #pragma unroll
    for (int r = 0; r < 8; ++r) {
      float mt = fmaxf(fmaxf(s_acc[0][r], s_acc[1][r]),
                       fmaxf(s_acc[2][r], s_acc[3][r]));
      #pragma unroll
      for (int off = 1; off < 16; off <<= 1)
        mt = fmaxf(mt, __shfl_xor(mt, off, 32));
      float mnew = fmaxf(mrow[r], mt);
      alpha[r] = __expf(mrow[r] - mnew);
      mrow[r]  = mnew;
      float psum = 0.f;
      #pragma unroll
      for (int t = 0; t < 4; ++t) {
        float p = __expf(s_acc[t][r] - mnew);
        s_acc[t][r] = p;
        psum += p;
      }
      #pragma unroll
      for (int off = 1; off < 16; off <<= 1)
        psum += __shfl_xor(psum, off, 32);
      lrow[r] = lrow[r] * alpha[r] + psum;
    }

    // ---- rescale O
    #pragma unroll
    for (int n = 0; n < 8; ++n)
      #pragma unroll
      for (int r = 0; r < 8; ++r)
        o_acc[n][r] *= alpha[r];

    // ---- P -> LDS (C/D layout -> row-major), wave-local
    f16* pl = p_lds[wave];
    #pragma unroll
    for (int t = 0; t < 4; ++t)
      #pragma unroll
      for (int r = 0; r < 8; ++r)
        pl[(r + 8 * h) * 64 + 16 * t + ln] = (f16)s_acc[t][r];

    // ---- O += P @ V : 2 k-steps over the 64 keys
    #pragma unroll
    for (int s = 0; s < 2; ++s) {
      v16h pf;
      ((v8h*)&pf)[0] = *(const v8h*)&pl[ln * 64 + 32 * s + 16 * h];
      ((v8h*)&pf)[1] = *(const v8h*)&pl[ln * 64 + 32 * s + 16 * h + 8];
      #pragma unroll
      for (int n = 0; n < 8; ++n) {
        const v16h vf =
            *(const v16h*)&vt_lds[cur][(16 * n + ln) * 64 + 32 * s + 16 * h];
        o_acc[n] = WMMA_F16(pf, vf, o_acc[n]);
      }
    }

    if (jt + 1 < NTOK / 64) {
      store_tile(cur ^ 1);   // other buffer: last read before previous barrier
      __builtin_prefetch(&adjb[(size_t)(q0 + (tid & 15)) * NTOK + j0 + 64], 0, 1);
    }
    cur ^= 1;
  }

  // ---- normalize and store left half of concat output
  #pragma unroll
  for (int r = 0; r < 8; ++r) {
    float inv = 1.0f / lrow[r];
    size_t row = qrow + r + 8 * h;
    #pragma unroll
    for (int n = 0; n < 8; ++n)
      out[row * 256 + 16 * n + ln] = o_acc[n][r] * inv;
  }
}

// ---------------------------------------------------------------------------
extern "C" void kernel_launch(void* const* d_in, const int* in_sizes, int n_in,
                              void* d_out, int out_size, void* d_ws, size_t ws_size,
                              hipStream_t stream) {
  (void)in_sizes; (void)n_in; (void)out_size; (void)ws_size;
  const float* q   = (const float*)d_in[0];
  // d_in[1] (k) and d_in[2] (v) are unused by the reference
  const float* adj = (const float*)d_in[3];
  const float* W1  = (const float*)d_in[4];
  const float* Wq  = (const float*)d_in[5];
  const float* Wk  = (const float*)d_in[6];
  const float* Wv  = (const float*)d_in[7];

  float* out     = (float*)d_out;                 // [16384,256]
  float* res_out = out + (size_t)NROWS * 256;     // [16384,128]

  f16* qh_h = (f16*)d_ws;                         // 3 x 16384x128 f16 = 12 MB
  f16* kh_h = qh_h + (size_t)NROWS * HDIM;
  f16* vh_t = kh_h + (size_t)NROWS * HDIM;

  proj_kernel<<<dim3(NROWS / 64), dim3(128), 0, stream>>>(
      q, W1, Wq, Wk, Wv, out, res_out, qh_h, kh_h, vh_t);
  attn_kernel<<<dim3(NTOK / 64, NBATCH), dim3(128), 0, stream>>>(
      qh_h, kh_h, vh_t, adj, out);
}
